// PointNetFeaturePropagation_87488483819933
// MI455X (gfx1250) — matrix operand order repaired and checked by reading, not compiled, not run on hardware
//
#include <hip/hip_runtime.h>
#include <hip/hip_bf16.h>

typedef _Float16 half_t;
typedef __attribute__((ext_vector_type(16))) _Float16 v16h;
typedef __attribute__((ext_vector_type(8)))  float    v8f;
typedef __attribute__((ext_vector_type(4)))  uint32_t u32x4;
typedef __attribute__((ext_vector_type(8)))  uint32_t u32x8;

struct alignas(16) H8 { _Float16 h[8]; };

#define B_   8
#define N_   8192
#define S_   2048
#define D1_  128
#define D2_  256
#define J_   (B_ * N_)       // 65536 total points
#define CH_  256             // output channels (both layers)
#define K1_  384
#define K2_  256

// ---------------- workspace layout (bytes) ----------------
#define OFF_W0H    (size_t)0                       // 256*384*2   = 196608
#define OFF_W1H    (size_t)196608                  // 256*256*2   = 131072
#define OFF_P2T    (size_t)327680                  // 8*2048*256*2 = 8388608
#define OFF_IDX    (size_t)8716288                 // 65536*3*4   = 786432
#define OFF_WGT    (size_t)9502720                 // 65536*3*4   = 786432
#define OFF_INTERP (size_t)10289152                // 65536*256*2 = 33554432 (reused as X2 row-major)
#define OFF_Y      (size_t)43843584                // 65536*256*4 = 67108864 (Y1 then Y2)
#define OFF_STATS  (size_t)110952448               // 2 * (256 mean + 256 rstd) * 4

// ---------------- small f32 -> f16 convert ----------------
__global__ void cvt_f16_kernel(const float* __restrict__ in, half_t* __restrict__ out, int n) {
    int i = blockIdx.x * blockDim.x + threadIdx.x;
    if (i < n) out[i] = (half_t)in[i];
}

// ---------------- points2 [B][256][2048] f32 -> p2t [B][2048][256] f16 ----------------
__global__ void transpose_p2_kernel(const float* __restrict__ p2, half_t* __restrict__ p2t) {
    __shared__ float tile[32][33];
    int b  = blockIdx.z;
    int s0 = blockIdx.x * 32;
    int c0 = blockIdx.y * 32;
    int tx = threadIdx.x, ty = threadIdx.y;
    for (int i = ty; i < 32; i += 8)
        tile[i][tx] = p2[((size_t)b * D2_ + c0 + i) * S_ + s0 + tx];
    __syncthreads();
    for (int i = ty; i < 32; i += 8)
        p2t[((size_t)b * S_ + s0 + i) * D2_ + c0 + tx] = (half_t)tile[tx][i];
}

// ---------------- 3-NN + inverse-distance weights ----------------
__global__ void __launch_bounds__(256) knn_kernel(const float* __restrict__ xyz1,
                                                  const float* __restrict__ xyz2,
                                                  int* __restrict__ idxb,
                                                  float* __restrict__ wb) {
    __shared__ float sx[S_], sy[S_], sz[S_], sq[S_];
    int b     = blockIdx.x >> 5;          // 32 blocks per batch (8192/256)
    int nBase = (blockIdx.x & 31) * 256;
    const float* base2 = xyz2 + (size_t)b * 3 * S_;
    for (int i = threadIdx.x; i < S_; i += 256) {
        sx[i] = base2[i];
        sy[i] = base2[S_ + i];
        sz[i] = base2[2 * S_ + i];
    }
    __syncthreads();
    for (int i = threadIdx.x; i < S_; i += 256)
        sq[i] = sx[i] * sx[i] + sy[i] * sy[i] + sz[i] * sz[i];
    __syncthreads();

    int n = nBase + threadIdx.x;
    const float* base1 = xyz1 + (size_t)b * 3 * N_;
    float x = base1[n], y = base1[N_ + n], z = base1[2 * N_ + n];
    float q = x * x + y * y + z * z;

    float d0 = 3.4e38f, d1 = 3.4e38f, d2v = 3.4e38f;
    int   i0 = 0, i1 = 0, i2 = 0;
    for (int s = 0; s < S_; ++s) {
        // reference formula: |a|^2 + |b|^2 - 2 a.b
        float d = q + sq[s] - 2.f * (x * sx[s] + y * sy[s] + z * sz[s]);
        if (d < d2v) {
            if (d < d1) {
                d2v = d1; i2 = i1;
                if (d < d0) { d1 = d0; i1 = i0; d0 = d; i0 = s; }
                else        { d1 = d;  i1 = s; }
            } else { d2v = d; i2 = s; }
        }
    }
    int j = b * N_ + n;
    float r0 = 1.f / (d0 + 1e-8f), r1 = 1.f / (d1 + 1e-8f), r2 = 1.f / (d2v + 1e-8f);
    float rs = 1.f / (r0 + r1 + r2);
    idxb[j * 3 + 0] = i0; idxb[j * 3 + 1] = i1; idxb[j * 3 + 2] = i2;
    wb[j * 3 + 0] = r0 * rs; wb[j * 3 + 1] = r1 * rs; wb[j * 3 + 2] = r2 * rs;
}

// ---------------- weighted gather: interp [J][256] f16 ----------------
__global__ void __launch_bounds__(256) interp_kernel(const half_t* __restrict__ p2t,
                                                     const int* __restrict__ idxb,
                                                     const float* __restrict__ wb,
                                                     half_t* __restrict__ interp) {
    int wave = threadIdx.x >> 5;
    int lane = threadIdx.x & 31;
    int j = blockIdx.x * 8 + wave;      // one wave32 per point
    int b = j >> 13;
    int i0 = idxb[j * 3 + 0], i1 = idxb[j * 3 + 1], i2 = idxb[j * 3 + 2];
    float w0 = wb[j * 3 + 0], w1 = wb[j * 3 + 1], w2 = wb[j * 3 + 2];
    const H8* r0 = (const H8*)(p2t + ((size_t)b * S_ + i0) * D2_) + lane;
    const H8* r1 = (const H8*)(p2t + ((size_t)b * S_ + i1) * D2_) + lane;
    const H8* r2 = (const H8*)(p2t + ((size_t)b * S_ + i2) * D2_) + lane;
    H8 a = *r0, c = *r1, e = *r2, o;
    for (int t = 0; t < 8; ++t)
        o.h[t] = (half_t)(w0 * (float)a.h[t] + w1 * (float)c.h[t] + w2 * (float)e.h[t]);
    ((H8*)(interp + (size_t)j * D2_))[lane] = o;
}

// ---------------- WMMA GEMM:  Y[o][j] = sum_k W[o][k] * X[k][j] ----------------
// Block: 256 threads = 8 waves (2 M x 4 N); tile 128(M) x 128(J); K-step 32.
// Row-major f16 K-tiles are staged by the Tensor Data Mover (one tensor_load_to_lds
// per K-step, with D# padding reproducing the [128][40] bank-padded LDS layout).
__global__ void __launch_bounds__(256) gemm_kernel(const half_t* __restrict__ W,
                                                   const float*  __restrict__ srcF,   // k-major f32 (points1) for kb < Ktrans
                                                   const half_t* __restrict__ srcRow, // row-major f16 rows of 256 (interp / X2)
                                                   float* __restrict__ Y,
                                                   int K, int Ktrans) {
    __shared__ half_t Xs[128][40];      // [j][k], +8 halves pad; row stride 80B (16B aligned)
    const int t      = threadIdx.x;
    const int waveId = t >> 5;
    const int lane   = t & 31;
    const int lrow   = lane & 15;
    const int hi     = lane >> 4;       // lane half selects contiguous K sub-range 16*hi..16*hi+15
    const int wm     = waveId >> 2;     // 0..1
    const int wn     = waveId & 3;      // 0..3
    const int jBase  = blockIdx.x * 128;
    const int mBase  = blockIdx.y * 128;
    const int bBlk   = jBase >> 13;     // 128-column stripe stays inside one batch
    const int nBase  = jBase & (N_ - 1);

    v8f acc[4][2];
    for (int mf = 0; mf < 4; ++mf)
        for (int nf = 0; nf < 2; ++nf)
            acc[mf][nf] = {};

    for (int kb = 0; kb < K; kb += 32) {
        // ---- stage X tile [32 x 128] transposed into Xs[j][k] ----
        if (kb < Ktrans) {                         // f32, [channel][n] layout (points1): manual transpose
            for (int i = 0; i < 16; ++i) {
                int lin = i * 256 + t;
                int k = lin >> 7, j = lin & 127;   // consecutive t -> consecutive j (coalesced)
                float v = srcF[((size_t)(bBlk * D1_ + kb + k)) * N_ + nBase + j];
                Xs[j][k] = (half_t)v;
            }
        } else if (waveId == 0) {
            // ---- TDM: DMA 128 rows x 32 halves from global into padded LDS tile ----
            const half_t* gp = srcRow + ((size_t)jBase) * 256 + (kb - Ktrans);
            uint64_t ga = (uint64_t)(uintptr_t)gp;
            uint32_t ldsAddr = (uint32_t)(uintptr_t)(&Xs[0][0]);   // flat low 32 bits = LDS offset
            u32x4 g0;
            g0[0] = 1u;                                            // count=1, user descriptor, no gather
            g0[1] = ldsAddr;                                       // D#.lds_addr
            g0[2] = (uint32_t)ga;                                  // D#.global_addr[31:0]
            g0[3] = (uint32_t)((ga >> 32) & 0x1FFFFFFull)          // D#.global_addr[56:32]
                  | (2u << 30);                                    // D#.type = 2 (image)
            u32x8 g1;
            g1[0] = (1u << 16)                                     // data_size = 2 bytes
                  | (1u << 20)                                     // pad_enable
                  | (3u << 22)                                     // pad_interval: 16 DWORDs (= 32 halves)
                  | (3u << 25);                                    // pad_amount:   4 DWORDs (=  8 halves)
            g1[1] = (256u & 0xFFFFu) << 16;                        // tensor_dim0[15:0] = 256
            g1[2] = (256u >> 16) | ((65536u & 0xFFFFu) << 16);     // tensor_dim0 hi | tensor_dim1 lo
            g1[3] = (65536u >> 16) | (32u << 16);                  // tensor_dim1 hi | tile_dim0 = 32
            g1[4] = 128u;                                          // tile_dim1 = 128, tile_dim2 = 0
            g1[5] = 256u;                                          // tensor_dim0_stride = 256 elements
            g1[6] = 0u;
            g1[7] = 0u;
            asm volatile("tensor_load_to_lds %0, %1" :: "s"(g0), "s"(g1) : "memory");
            __builtin_amdgcn_s_wait_tensorcnt(0);
        }
        __syncthreads();

        // ---- A fragments straight from global W (rows contiguous in K) ----
        // A loaded in contiguous K order (same as B): lanes 0-15 hold K {0..15}, lanes 16-31 hold K {16..31}
        union Frag { v16h v; H8 p[2]; };
        Frag a[4], bf[2];
        for (int mf = 0; mf < 4; ++mf) {
            int r = mBase + wm * 64 + mf * 16 + lrow;
            const half_t* wp = W + (size_t)r * K + kb + hi * 16;
            a[mf].p[0] = *(const H8*)wp;
            a[mf].p[1] = *(const H8*)(wp + 8);
        }
        // ---- B fragments from LDS: lane = column, 16 contiguous K halves per lane half ----
        for (int nf = 0; nf < 2; ++nf) {
            int c = wn * 32 + nf * 16 + lrow;
            const H8* xp = (const H8*)&Xs[c][hi * 16];
            bf[nf].p[0] = xp[0];
            bf[nf].p[1] = xp[1];
        }
        if (kb + 32 < K)   // prefetch next weight row chunk (global_prefetch_b8)
            __builtin_prefetch(W + (size_t)(mBase + wm * 64 + lrow) * K + kb + 32, 0, 3);

        for (int mf = 0; mf < 4; ++mf)
            for (int nf = 0; nf < 2; ++nf)
                acc[mf][nf] = __builtin_amdgcn_wmma_f32_16x16x32_f16(
                    false, a[mf].v, false, bf[nf].v, (short)0, acc[mf][nf], false, false);
        __syncthreads();
    }

    // ---- epilogue: C/D layout -> channel-major coalesced stores ----
    for (int mf = 0; mf < 4; ++mf)
        for (int nf = 0; nf < 2; ++nf)
            for (int v = 0; v < 8; ++v) {
                int o = mBase + wm * 64 + mf * 16 + v + hi * 8;
                int j = jBase + wn * 32 + nf * 16 + lrow;
                Y[(size_t)o * J_ + j] = acc[mf][nf][v];
            }
}

// ---------------- per-channel BN stats: one block per channel ----------------
__global__ void __launch_bounds__(256) bn_stats_kernel(const float* __restrict__ Y,
                                                       float* __restrict__ stats) {
    __shared__ float s1[256], s2[256];
    int o = blockIdx.x;
    const float* row = Y + (size_t)o * J_;
    float a = 0.f, b = 0.f;
    for (int i = threadIdx.x; i < J_; i += 256) {
        float v = row[i];
        a += v;
        b += v * v;
    }
    s1[threadIdx.x] = a; s2[threadIdx.x] = b;
    __syncthreads();
    for (int st = 128; st > 0; st >>= 1) {
        if (threadIdx.x < st) {
            s1[threadIdx.x] += s1[threadIdx.x + st];
            s2[threadIdx.x] += s2[threadIdx.x + st];
        }
        __syncthreads();
    }
    if (threadIdx.x == 0) {
        float m   = s1[0] * (1.f / J_);
        float var = s2[0] * (1.f / J_) - m * m;
        stats[o]        = m;
        stats[256 + o]  = rsqrtf(var + 1e-5f);
    }
}

// ---------------- BN + ReLU + transpose -> X2 row-major [J][256] f16 ----------------
__global__ void bn_relu_tr_kernel(const float* __restrict__ Y, const float* __restrict__ stats,
                                  const float* __restrict__ g, const float* __restrict__ beta,
                                  half_t* __restrict__ X2) {
    __shared__ half_t tile[32][33];
    int j0 = blockIdx.x * 32;
    int o0 = blockIdx.y * 32;
    int tx = threadIdx.x, ty = threadIdx.y;
    for (int i = ty; i < 32; i += 8) {
        int o = o0 + i;
        float v = Y[(size_t)o * J_ + j0 + tx];
        float r = (v - stats[o]) * stats[256 + o] * g[o] + beta[o];
        tile[i][tx] = (half_t)fmaxf(r, 0.f);
    }
    __syncthreads();
    for (int i = ty; i < 32; i += 8)
        X2[(size_t)(j0 + i) * 256 + o0 + tx] = tile[tx][i];
}

// ---------------- BN + ReLU -> f32 output [B][256][N] ----------------
__global__ void bn_relu_out_kernel(const float* __restrict__ Y, const float* __restrict__ stats,
                                   const float* __restrict__ g, const float* __restrict__ beta,
                                   float* __restrict__ out) {
    size_t idx = (size_t)blockIdx.x * blockDim.x + threadIdx.x;
    int o = (int)(idx >> 16);
    int j = (int)(idx & 65535);
    int b = j >> 13;
    int n = j & (N_ - 1);
    float v = Y[idx];
    float r = (v - stats[o]) * stats[256 + o] * g[o] + beta[o];
    out[((size_t)b * CH_ + o) * N_ + n] = fmaxf(r, 0.f);
}

extern "C" void kernel_launch(void* const* d_in, const int* in_sizes, int n_in,
                              void* d_out, int out_size, void* d_ws, size_t ws_size,
                              hipStream_t stream) {
    const float* xyz1    = (const float*)d_in[0];
    const float* xyz2    = (const float*)d_in[1];
    const float* points1 = (const float*)d_in[2];
    const float* points2 = (const float*)d_in[3];
    const float* W0      = (const float*)d_in[4];
    // b0/b1 (d_in[5], d_in[9]) cancel under training-mode BN (mean subtraction) -> skipped
    const float* g0      = (const float*)d_in[6];
    const float* beta0   = (const float*)d_in[7];
    const float* W1      = (const float*)d_in[8];
    const float* g1      = (const float*)d_in[10];
    const float* beta1   = (const float*)d_in[11];

    char* ws = (char*)d_ws;
    half_t* W0h    = (half_t*)(ws + OFF_W0H);
    half_t* W1h    = (half_t*)(ws + OFF_W1H);
    half_t* p2t    = (half_t*)(ws + OFF_P2T);
    int*    idxb   = (int*)   (ws + OFF_IDX);
    float*  wgt    = (float*) (ws + OFF_WGT);
    half_t* interp = (half_t*)(ws + OFF_INTERP);
    half_t* X2     = (half_t*)(ws + OFF_INTERP);   // overlays interp (dead after GEMM1)
    float*  Y      = (float*) (ws + OFF_Y);        // Y1, then Y2 (Y1 dead after BN-apply1)
    float*  stats1 = (float*) (ws + OFF_STATS);
    float*  stats2 = (float*) (ws + OFF_STATS + 2048);

    // 1) weights -> f16
    cvt_f16_kernel<<<(CH_ * K1_ + 255) / 256, 256, 0, stream>>>(W0, W0h, CH_ * K1_);
    cvt_f16_kernel<<<(CH_ * K2_ + 255) / 256, 256, 0, stream>>>(W1, W1h, CH_ * K2_);

    // 2) points2 -> [B][S][D2] f16
    transpose_p2_kernel<<<dim3(S_ / 32, D2_ / 32, B_), dim3(32, 8), 0, stream>>>(points2, p2t);

    // 3) 3-NN + weights
    knn_kernel<<<J_ / 256, 256, 0, stream>>>(xyz1, xyz2, idxb, wgt);

    // 4) weighted feature gather
    interp_kernel<<<J_ / 8, 256, 0, stream>>>(p2t, idxb, wgt, interp);

    // 5) layer 1 GEMM: K = 128 (points1, f32 k-major) + 256 (interp via TDM)
    gemm_kernel<<<dim3(J_ / 128, CH_ / 128), 256, 0, stream>>>(
        W0h, points1, interp, Y, K1_, D1_);

    // 6) BN stats + apply + ReLU + transpose -> X2 row-major f16
    bn_stats_kernel<<<CH_, 256, 0, stream>>>(Y, stats1);
    bn_relu_tr_kernel<<<dim3(J_ / 32, CH_ / 32), dim3(32, 8), 0, stream>>>(Y, stats1, g0, beta0, X2);

    // 7) layer 2 GEMM (X2 row-major f16 via TDM) -> Y2
    gemm_kernel<<<dim3(J_ / 128, CH_ / 128), 256, 0, stream>>>(
        W1h, nullptr, X2, Y, K2_, 0);

    // 8) BN stats + apply + ReLU -> d_out [B][256][N] f32
    bn_stats_kernel<<<CH_, 256, 0, stream>>>(Y, stats2);
    bn_relu_out_kernel<<<(CH_ * J_) / 256, 256, 0, stream>>>(Y, stats2, g1, beta1, (float*)d_out);
}
